// LPKTNet_54786602827936
// MI455X (gfx1250) — matrix operand order repaired and checked by reading, not compiled, not run on hardware
//
#include <hip/hip_runtime.h>

#define B_        64
#define S_        200
#define DK        32
#define NQ1       4097
#define NTILES    257            // ceil(4097/16)
#define NFULL     256            // tiles with no guards needed
#define NROWS_PAD (NTILES * 16)  // 4112
#define HSTR      36             // padded f16 row stride (72B, 8B aligned)

typedef __attribute__((ext_vector_type(16))) _Float16 v16h;
typedef __attribute__((ext_vector_type(4)))  _Float16 h16x4;
typedef __attribute__((ext_vector_type(8)))  float    v8f;

// fast sigmoid: v_exp_f32 + v_rcp_f32 (no IEEE divide expansion)
__device__ __forceinline__ float sigm(float x) {
    return __builtin_amdgcn_rcpf(1.0f + __expf(-x));
}

extern "C" __global__ void __launch_bounds__(256, 1)
lpkt_kernel(const int* __restrict__ e_data, const int* __restrict__ at_data,
            const int* __restrict__ it_data, const float* __restrict__ a_data,
            const float* __restrict__ qmat,
            const float* __restrict__ Ee, const float* __restrict__ Eat,
            const float* __restrict__ Eit,
            const float* __restrict__ W1, const float* __restrict__ b1,
            const float* __restrict__ W2, const float* __restrict__ b2,
            const float* __restrict__ W3, const float* __restrict__ b3,
            const float* __restrict__ W4, const float* __restrict__ b4,
            const float* __restrict__ W5, const float* __restrict__ b5,
            const float* __restrict__ h0, float* __restrict__ out)
{
    extern __shared__ char smem[];
    _Float16* hS = (_Float16*)smem;                                  // 4112 x 36 f16
    float* fS    = (float*)(smem + (size_t)NROWS_PAD * HSTR * 2);
    float* xS    = fS;          // 128 : [learning_pre | it | learning | h_tilde_pre]
    float* LGs   = fS + 128;    // 32
    float* cS    = fS + 160;    // 32
    float* htS   = fS + 192;    // 32 h_tilde
    float* haccS = fS + 224;    // 32 h_tilde accumulator
    float* lpS   = fS + 256;    // 32 learning_pre
    float* lnS   = fS + 288;    // 32 learning
    float* lacc  = fS + 320;    // 32 learning accumulator
    float* s2a   = fS + 352;    // 32 x@W2 accumulator
    float* s3a   = fS + 384;    // 32 x@W3 accumulator

    const int b    = blockIdx.x;
    const int tid  = threadIdx.x;
    const int lane = tid & 31;
    const int wave = tid >> 5;          // 8 waves
    const int grp  = tid >> 5;
    const int dcol = tid & 31;

    // per-lane layout constants (operand-swapped scheme)
    const int nlane = lane & 15;          // row-in-tile this lane owns in B/D
    const int khalf = lane >> 4;          // 0: k/d-low half, 1: high half
    const int ka    = khalf * 8;          // C/D row base (acc element r -> m = ka + r)
    const int kb    = khalf * 16;         // A/B k-base (contiguous k = kb + s)
    const int dB    = khalf * 24;         // d-base of chunk living in B regs (0 or 24)
    const int dL    = 16 - khalf * 8;     // d-base of extra-loaded chunk (16 or 8)

    // ---- load h0 -> LDS as f16; zero pad rows ----
    for (int i = tid; i < NQ1 * DK; i += 256) {
        int r = i >> 5, c = i & 31;
        hS[r * HSTR + c] = (_Float16)h0[i];
    }
    for (int i = tid; i < (NROWS_PAD - NQ1) * HSTR; i += 256) {
        int r = NQ1 + i / HSTR, c = i % HSTR;
        hS[r * HSTR + c] = (_Float16)0.f;
    }
    if (tid < 32) { haccS[tid] = 0.f; lpS[tid] = 0.f; }
    if (tid == 0) out[b * S_] = 0.f;
    __syncthreads();

    // ---- initial h_tilde0 = sum_n q[e0][n] * h0[n,:] ----
    {
        const float* q0 = qmat + (long)e_data[b * S_] * NQ1;
        float acc = 0.f;
        for (int r = grp; r < NQ1; r += 8)
            acc += q0[r] * (float)hS[r * HSTR + dcol];
        atomicAdd(&haccS[dcol], acc);
    }
    __syncthreads();
    if (tid < 32) htS[tid] = haccS[tid];
    __syncthreads();

    // ---- preload A operands: W4h row tiles (contiguous k = 16h + s), step-invariant ----
    v16h A0, A1;
    {
        int m = nlane;
        #pragma unroll
        for (int i = 0; i < 8; ++i) {
            A0[i]     = (_Float16)W4[(m)      * 128 + kb + i];        // K = kb+i
            A0[8 + i] = (_Float16)W4[(m)      * 128 + kb + 8 + i];    // K = kb+8+i
            A1[i]     = (_Float16)W4[(m + 16) * 128 + kb + i];
            A1[8 + i] = (_Float16)W4[(m + 16) * 128 + kb + 8 + i];
        }
    }

    for (int t = 0; t < S_ - 1; ++t) {
        const int   et  = e_data[b * S_ + t];
        const int   enx = e_data[b * S_ + t + 1];
        const int   att = at_data[b * S_ + t];
        const int   itt = it_data[b * S_ + t];
        const float av  = a_data[b * S_ + t];
        const float* qe = qmat + (long)et  * NQ1;
        const float* qn = qmat + (long)enx * NQ1;

        // ---- P0: init accumulators + ready x pieces ----
        if (tid < 32) {
            lacc[tid]  = b1[tid];
            s2a[tid]   = b2[tid];
            s3a[tid]   = b3[tid];
            haccS[tid] = 0.f;
            xS[tid]      = lpS[tid];
            xS[32 + tid] = Eit[itt * 32 + tid];
            xS[96 + tid] = htS[tid];
        }
        __syncthreads();

        // ---- P1: learning partials (96 features, 12 per group) ----
        {
            float s = 0.f;
            #pragma unroll
            for (int j = 0; j < 12; ++j) {
                int f = grp * 12 + j;
                float xv = (f < 32) ? Ee[et * 32 + f]
                         : (f < 64) ? Eat[att * 32 + (f - 32)]
                                    : av;
                s += xv * W1[dcol * 96 + f];
            }
            atomicAdd(&lacc[dcol], s);
        }
        __syncthreads();

        // ---- P2: commit learning into x ----
        if (tid < 32) { float l = lacc[tid]; lnS[tid] = l; xS[64 + tid] = l; }
        __syncthreads();

        // ---- P3: x@W2 / x@W3 partials ----
        {
            float s2 = 0.f, s3 = 0.f;
            #pragma unroll
            for (int j = 0; j < 16; ++j) {
                int k = grp * 16 + j;
                float xv = xS[k];
                s2 += xv * W2[dcol * 128 + k];
                s3 += xv * W3[dcol * 128 + k];
            }
            atomicAdd(&s2a[dcol], s2);
            atomicAdd(&s3a[dcol], s3);
        }
        __syncthreads();

        // ---- P4: LG ----
        if (tid < 32) {
            float lg = tanhf(s2a[tid]);
            float gl = sigm(s3a[tid]);
            LGs[tid] = gl * (lg + 1.f) * 0.5f;
        }
        __syncthreads();

        // ---- P5: c = LG@W4l.T + it@W4i.T + b4 ----
        if (tid < 32) {
            float s = b4[tid];
            for (int k = 0; k < 32; ++k) {
                s += LGs[k]     * W4[tid * 128 + 32 + k];
                s += xS[32 + k] * W4[tid * 128 + 64 + k];
            }
            cS[tid] = s;
        }
        __syncthreads();

        // ---- step-constant per-lane vectors ----
        v8f C0, C1;                 // C[h][r] = c[16h + ka + r]
        float lgB[8], lgL[8];       // LG by output chunk
        #pragma unroll
        for (int r = 0; r < 8; ++r) {
            C0[r]  = cS[ka + r];
            C1[r]  = cS[16 + ka + r];
            lgB[r] = LGs[dB + r];
            lgL[r] = LGs[dL + r];
        }

        float accB[8], accL[8];
        #pragma unroll
        for (int r = 0; r < 8; ++r) { accB[r] = 0.f; accL[r] = 0.f; }

        // ---- tile sweep (tiles 0..255: no guards) ----
        for (int rt = wave; rt < NFULL; rt += 8) {
            const int nrow = rt * 16 + nlane;
            _Float16* hrow = hS + nrow * HSTR;

            // B operand: 16 contiguous f16 of this lane's h row
            const h16x4* bp = (const h16x4*)(hrow + kb);
            h16x4 b0 = bp[0], b1v = bp[1], b2v = bp[2], b3v = bp[3];
            v16h Bv;
            #pragma unroll
            for (int i = 0; i < 4; ++i) {
                Bv[i] = b0[i]; Bv[4+i] = b1v[i]; Bv[8+i] = b2v[i]; Bv[12+i] = b3v[i];
            }
            // extra h_pre chunk not covered by B regs
            const h16x4* lp = (const h16x4*)(hrow + dL);
            h16x4 l0 = lp[0], l1 = lp[1];
            // h_pre chunk that lives inside B regs
            h16x4 hb0 = khalf ? b2v : b0;
            h16x4 hb1 = khalf ? b3v : b1v;

            float qev = qe[nrow];
            float qnv = qn[nrow];

            v8f D0 = __builtin_amdgcn_wmma_f32_16x16x32_f16(
                         false, A0, false, Bv, (short)0, C0, false, false);
            v8f D1 = __builtin_amdgcn_wmma_f32_16x16x32_f16(
                         false, A1, false, Bv, (short)0, C1, false, false);

            // route D regs to this lane's chunks
            v8f DB = khalf ? D1 : D0;
            v8f DL = khalf ? D0 : D1;

            h16x4 oB0, oB1, oL0, oL1;
            #pragma unroll
            for (int r = 0; r < 8; ++r) {
                float hpB = (float)(r < 4 ? hb0[r & 3] : hb1[r & 3]);
                float hpL = (float)(r < 4 ? l0[r & 3]  : l1[r & 3]);
                float hnB = qev * lgB[r] + sigm(DB[r]) * hpB;
                float hnL = qev * lgL[r] + sigm(DL[r]) * hpL;
                if (r < 4) { oB0[r & 3] = (_Float16)hnB; oL0[r & 3] = (_Float16)hnL; }
                else       { oB1[r & 3] = (_Float16)hnB; oL1[r & 3] = (_Float16)hnL; }
                accB[r] += qnv * hnB;
                accL[r] += qnv * hnL;
            }
            h16x4* sb = (h16x4*)(hrow + dB);
            h16x4* sl = (h16x4*)(hrow + dL);
            sb[0] = oB0; sb[1] = oB1;
            sl[0] = oL0; sl[1] = oL1;
        }

        // ---- tail tile 256 (rows 4096..4111; only n==4096 valid), wave 0 ----
        if (wave == 0) {
            const int nrow = NFULL * 16 + nlane;
            const bool valid = nrow < NQ1;
            _Float16* hrow = hS + nrow * HSTR;

            const h16x4* bp = (const h16x4*)(hrow + kb);
            h16x4 b0 = bp[0], b1v = bp[1], b2v = bp[2], b3v = bp[3];
            v16h Bv;
            #pragma unroll
            for (int i = 0; i < 4; ++i) {
                Bv[i] = b0[i]; Bv[4+i] = b1v[i]; Bv[8+i] = b2v[i]; Bv[12+i] = b3v[i];
            }
            const h16x4* lp = (const h16x4*)(hrow + dL);
            h16x4 l0 = lp[0], l1 = lp[1];
            h16x4 hb0 = khalf ? b2v : b0;
            h16x4 hb1 = khalf ? b3v : b1v;

            float qev = valid ? qe[nrow] : 0.f;
            float qnv = valid ? qn[nrow] : 0.f;

            v8f D0 = __builtin_amdgcn_wmma_f32_16x16x32_f16(
                         false, A0, false, Bv, (short)0, C0, false, false);
            v8f D1 = __builtin_amdgcn_wmma_f32_16x16x32_f16(
                         false, A1, false, Bv, (short)0, C1, false, false);
            v8f DB = khalf ? D1 : D0;
            v8f DL = khalf ? D0 : D1;

            h16x4 oB0, oB1, oL0, oL1;
            #pragma unroll
            for (int r = 0; r < 8; ++r) {
                float hpB = (float)(r < 4 ? hb0[r & 3] : hb1[r & 3]);
                float hpL = (float)(r < 4 ? l0[r & 3]  : l1[r & 3]);
                float hnB = qev * lgB[r] + sigm(DB[r]) * hpB;
                float hnL = qev * lgL[r] + sigm(DL[r]) * hpL;
                if (r < 4) { oB0[r & 3] = (_Float16)hnB; oL0[r & 3] = (_Float16)hnL; }
                else       { oB1[r & 3] = (_Float16)hnB; oL1[r & 3] = (_Float16)hnL; }
                accB[r] += qnv * hnB;
                accL[r] += qnv * hnL;
            }
            if (valid) {
                h16x4* sb = (h16x4*)(hrow + dB);
                h16x4* sl = (h16x4*)(hrow + dL);
                sb[0] = oB0; sb[1] = oB1;
                sl[0] = oL0; sl[1] = oL1;
            }
        }

        // ---- h_tilde reduction: butterfly within 16-lane groups, then LDS ----
        #pragma unroll
        for (int r = 0; r < 8; ++r) {
            #pragma unroll
            for (int off = 8; off >= 1; off >>= 1) {
                accB[r] += __shfl_xor(accB[r], off);
                accL[r] += __shfl_xor(accL[r], off);
            }
        }
        if (nlane == 0) {
            #pragma unroll
            for (int r = 0; r < 8; ++r) {
                atomicAdd(&haccS[dB + r], accB[r]);
                atomicAdd(&haccS[dL + r], accL[r]);
            }
        }
        // reset local accumulators for next step
        #pragma unroll
        for (int r = 0; r < 8; ++r) { accB[r] = 0.f; accL[r] = 0.f; }
        __syncthreads();

        // ---- postamble: commit h_tilde, learning_pre, emit y ----
        if (tid < 32) {
            htS[tid] = haccS[tid];
            lpS[tid] = lnS[tid];
        }
        __syncthreads();
        if (tid < 32) {
            int j = tid;
            float s = b5[j];
            for (int k = 0; k < 32; ++k) {
                s += Ee[enx * 32 + k] * W5[j * 64 + k];
                s += htS[k]           * W5[j * 64 + 32 + k];
            }
            float sg = sigm(s);
            for (int off = 16; off; off >>= 1) sg += __shfl_xor(sg, off);
            if (j == 0) out[b * S_ + t + 1] = sg * (1.f / 32.f);
        }
        __syncthreads();
    }
}

extern "C" void kernel_launch(void* const* d_in, const int* in_sizes, int n_in,
                              void* d_out, int out_size, void* d_ws, size_t ws_size,
                              hipStream_t stream) {
    (void)in_sizes; (void)n_in; (void)d_ws; (void)ws_size; (void)out_size;
    const int*   e_data = (const int*)  d_in[0];
    const int*   at_d   = (const int*)  d_in[1];
    const int*   it_d   = (const int*)  d_in[2];
    const float* a_d    = (const float*)d_in[3];
    const float* qmat   = (const float*)d_in[4];
    const float* Ee     = (const float*)d_in[5];
    const float* Eat    = (const float*)d_in[6];
    const float* Eit    = (const float*)d_in[7];
    const float* W1     = (const float*)d_in[8];
    const float* b1     = (const float*)d_in[9];
    const float* W2     = (const float*)d_in[10];
    const float* b2     = (const float*)d_in[11];
    const float* W3     = (const float*)d_in[12];
    const float* b3     = (const float*)d_in[13];
    const float* W4     = (const float*)d_in[14];
    const float* b4     = (const float*)d_in[15];
    const float* W5     = (const float*)d_in[16];
    const float* b5     = (const float*)d_in[17];
    const float* h0     = (const float*)d_in[18];
    float* out = (float*)d_out;

    size_t lds = (size_t)NROWS_PAD * HSTR * 2 + 448 * sizeof(float);
    lpkt_kernel<<<B_, 256, lds, stream>>>(e_data, at_d, it_d, a_d, qmat,
                                          Ee, Eat, Eit, W1, b1, W2, b2, W3, b3,
                                          W4, b4, W5, b5, h0, out);
}
